// HelmholtzGCNLayer_1864015806542
// MI455X (gfx1250) — matrix operand order repaired and checked by reading, not compiled, not run on hardware
//
#include <hip/hip_runtime.h>
#include <hip/hip_bf16.h>

typedef __attribute__((ext_vector_type(16))) __bf16 v16bf;
typedef __attribute__((ext_vector_type(8)))  float  v8f;

#define BN_EPS 1e-5f

__device__ __forceinline__ unsigned bfround(float f) {
    union { float f; unsigned u; } v; v.f = f;
    return v.u + 0x7FFFu + ((v.u >> 16) & 1u);         // round-to-nearest-even
}
__device__ __forceinline__ unsigned pack2bf(float lo, float hi) {
    return (bfround(hi) & 0xFFFF0000u) | (bfround(lo) >> 16);
}
__device__ __forceinline__ __bf16 f2bf(float f) {
    union { unsigned short s; __bf16 b; } o;
    o.s = (unsigned short)(bfround(f) >> 16);
    return o.b;
}

// ---------------- degree / normalization ----------------
__global__ void k_deg_init(float* deg, int n) {
    int i = blockIdx.x * blockDim.x + threadIdx.x;
    if (i < n) deg[i] = 1.0f;                 // self loop contributes 1
}
__global__ void k_deg_acc(const int* __restrict__ dst, float* deg, int e) {
    int i = blockIdx.x * blockDim.x + threadIdx.x;
    if (i < e) atomicAdd(&deg[dst[i]], 1.0f);
}
__global__ void k_rsqrt_inplace(float* deg, int n) {
    int i = blockIdx.x * blockDim.x + threadIdx.x;
    if (i < n) deg[i] = rsqrtf(deg[i]);       // deg -> dinv
}
__global__ void k_coef(const int* __restrict__ src, const int* __restrict__ dst,
                       const float* __restrict__ dinv, float* __restrict__ coef, int e) {
    int i = blockIdx.x * blockDim.x + threadIdx.x;
    if (i < e) coef[i] = dinv[src[i]] * dinv[dst[i]];
}

// ---------------- fp32 -> bf16 row-major streaming convert ----------------
__global__ void k_cvt_bf16(const float* __restrict__ in, unsigned* __restrict__ out,
                           long long npairs) {
    long long t = (long long)blockIdx.x * blockDim.x + threadIdx.x;
    if (t >= npairs) return;
    float2 f = ((const float2*)in)[t];
    out[t] = pack2bf(f.x, f.y);
}

// ---------------- pack W into per-lane WMMA B-fragment layout ----------------
// Fragment (kk, nt): 32 lanes x 16 bf16 (32B per lane, contiguous per fragment).
// Lane: n = lane&15, half = lane>>4; element i corresponds to K = kk*32 + half*16 + i.
template<int KT, int NT, int NCOLS>
__global__ void k_pack_w(const float* __restrict__ W, __bf16* __restrict__ out) {
    int t = blockIdx.x * blockDim.x + threadIdx.x;
    if (t >= KT * NT * 32) return;
    int kk   = t / (NT * 32);
    int nt   = (t / 32) % NT;
    int lane = t & 31;
    int n = lane & 15, half = lane >> 4;
    int col = nt * 16 + n;
    __bf16* o = out + (size_t)t * 16;
    #pragma unroll
    for (int i = 0; i < 16; ++i) {
        int k = kk * 32 + half * 16 + i;
        float val = (col < NCOLS) ? W[(size_t)k * NCOLS + col] : 0.0f;
        o[i] = f2bf(val);
    }
}

// ---------------- WMMA GEMM1: bf16 [N x 128] * packed W1 -> h1 fp32 [N x 64] ----
__global__ void k_gemm1_wmma(const __bf16* __restrict__ A, const __bf16* __restrict__ Bpk,
                             float* __restrict__ Out, int nrows, int nwaves) {
    int wave = (int)((blockIdx.x * blockDim.x + threadIdx.x) >> 5);
    if (wave >= nwaves) return;               // uniform exit -> EXEC all 1s for WMMA
    int lane = threadIdx.x & 31;
    int m = lane & 15, half = lane >> 4;
    int row0 = wave * 16;
    int ra = row0 + m; if (ra >= nrows) ra = nrows - 1;
    const __bf16* arow = A + (size_t)ra * 128;

    v8f acc[4] = {};
    #pragma unroll
    for (int kk = 0; kk < 4; ++kk) {
        v16bf a;
        ((uint4*)&a)[0] = *(const uint4*)(arow + kk * 32 + half * 16);       // K0+half*16..+7
        ((uint4*)&a)[1] = *(const uint4*)(arow + kk * 32 + half * 16 + 8);   // +8..+15
        #pragma unroll
        for (int nt = 0; nt < 4; ++nt) {
            v16bf b = *(const v16bf*)(Bpk + (size_t)(((kk * 4 + nt) * 32 + lane)) * 16);
            acc[nt] = __builtin_amdgcn_wmma_f32_16x16x32_bf16(
                false, a, false, b, (short)0, acc[nt], false, false);
        }
    }
    #pragma unroll
    for (int nt = 0; nt < 4; ++nt) {
        int col = nt * 16 + m;
        #pragma unroll
        for (int v = 0; v < 8; ++v) {
            int row = row0 + v + 8 * half;
            if (row < nrows) Out[(size_t)row * 64 + col] = acc[nt][v];
        }
    }
}

// ---------------- WMMA GEMM2: bf16 [N x 64] * packed W2 -> h2 fp32 [N x 40] ----
__global__ void k_gemm2_wmma(const __bf16* __restrict__ A, const __bf16* __restrict__ Bpk,
                             float* __restrict__ Out, int nrows, int nwaves) {
    int wave = (int)((blockIdx.x * blockDim.x + threadIdx.x) >> 5);
    if (wave >= nwaves) return;
    int lane = threadIdx.x & 31;
    int m = lane & 15, half = lane >> 4;
    int row0 = wave * 16;
    int ra = row0 + m; if (ra >= nrows) ra = nrows - 1;
    const __bf16* arow = A + (size_t)ra * 64;

    v8f acc[3] = {};
    #pragma unroll
    for (int kk = 0; kk < 2; ++kk) {
        v16bf a;
        ((uint4*)&a)[0] = *(const uint4*)(arow + kk * 32 + half * 16);
        ((uint4*)&a)[1] = *(const uint4*)(arow + kk * 32 + half * 16 + 8);
        #pragma unroll
        for (int nt = 0; nt < 3; ++nt) {
            v16bf b = *(const v16bf*)(Bpk + (size_t)(((kk * 3 + nt) * 32 + lane)) * 16);
            acc[nt] = __builtin_amdgcn_wmma_f32_16x16x32_bf16(
                false, a, false, b, (short)0, acc[nt], false, false);
        }
    }
    #pragma unroll
    for (int nt = 0; nt < 3; ++nt) {
        int col = nt * 16 + m;
        if (col >= 40) continue;
        #pragma unroll
        for (int v = 0; v < 8; ++v) {
            int row = row0 + v + 8 * half;
            if (row < nrows) Out[(size_t)row * 40 + col] = acc[nt][v];
        }
    }
}

// ---------------- edge scatter: agg[dst] += coef * h[src] ----------------
template<int C>
__global__ void k_scatter(const int* __restrict__ src, const int* __restrict__ dst,
                          const float* __restrict__ coef, const float* __restrict__ h,
                          float* __restrict__ agg, long long total) {
    long long t = (long long)blockIdx.x * blockDim.x + threadIdx.x;
    if (t >= total) return;
    int e = (int)(t / C);
    int c = (int)(t - (long long)e * C);
    float val = coef[e] * h[(size_t)src[e] * C + c];
    atomicAdd(&agg[(size_t)dst[e] * C + c], val);
}

// ---- epilogue 1: self-loop + k2*h + b, helm loss, BN, tanh -> bf16 x1 ----
__global__ void k_epi1(const float* __restrict__ h, const float* __restrict__ agg,
                       const float* __restrict__ dinv, const float* __restrict__ bias,
                       const float* __restrict__ k2p, const float* __restrict__ g,
                       const float* __restrict__ be, const float* __restrict__ mm,
                       const float* __restrict__ vv, __bf16* __restrict__ x1,
                       float* __restrict__ loss, int n) {
    int t = blockIdx.x * blockDim.x + threadIdx.x;   // over n*64
    float rs = 0.0f;
    if (t < n * 64) {
        int i = t >> 6, c = t & 63;
        float hv = h[t];
        float di = dinv[i];
        float a  = agg[t] + di * di * hv;            // A_hat h (with self loop)
        float k2 = k2p[0];
        float op = a + k2 * hv + bias[c];
        float r  = a + (k2 - 1.0f) * hv;
        rs = r * r;
        float bn = (op - mm[c]) * rsqrtf(vv[c] + BN_EPS) * g[c] + be[c];
        x1[t] = f2bf(tanhf(bn));
    }
    #pragma unroll
    for (int o = 16; o > 0; o >>= 1) rs += __shfl_xor(rs, o, 32);
    if ((threadIdx.x & 31) == 0 && rs != 0.0f) atomicAdd(loss, rs);
}

// ---------------- epilogue 2: self-loop + k2*h + b, BN, tanh -> fp32 x2 ----
__global__ void k_epi2(const float* __restrict__ h, const float* __restrict__ agg,
                       const float* __restrict__ dinv, const float* __restrict__ bias,
                       const float* __restrict__ k2p, const float* __restrict__ g,
                       const float* __restrict__ be, const float* __restrict__ mm,
                       const float* __restrict__ vv, float* __restrict__ x2, int n) {
    int t = blockIdx.x * blockDim.x + threadIdx.x;   // over n*40
    if (t >= n * 40) return;
    int i = t / 40, c = t - i * 40;
    float hv = h[t];
    float di = dinv[i];
    float a  = agg[t] + di * di * hv;
    float op = a + k2p[0] * hv + bias[c];
    float bn = (op - mm[c]) * rsqrtf(vv[c] + BN_EPS) * g[c] + be[c];
    x2[t] = tanhf(bn);
}

// ---------------- gather + log_softmax over 40 classes ----------------
__global__ void k_logsoftmax(const float* __restrict__ x2, const int* __restrict__ bn,
                             float* __restrict__ out, int nb) {
    int t = blockIdx.x * blockDim.x + threadIdx.x;
    if (t >= nb) return;
    const float* row = x2 + (size_t)bn[t] * 40;
    float mx = -3.402823466e38f;
    #pragma unroll
    for (int c = 0; c < 40; ++c) mx = fmaxf(mx, row[c]);
    float s = 0.0f;
    #pragma unroll
    for (int c = 0; c < 40; ++c) s += __expf(row[c] - mx);
    float ls = __logf(s) + mx;
    float* o = out + (size_t)t * 40;
    #pragma unroll
    for (int c = 0; c < 40; ++c) o[c] = row[c] - ls;
}

__global__ void k_finalize_loss(const float* __restrict__ loss, float* __restrict__ out,
                                int out_idx, float inv_count) {
    if (blockIdx.x == 0 && threadIdx.x == 0) out[out_idx] = loss[0] * inv_count;
}

extern "C" void kernel_launch(void* const* d_in, const int* in_sizes, int n_in,
                              void* d_out, int out_size, void* d_ws, size_t ws_size,
                              hipStream_t stream) {
    const float* features = (const float*)d_in[0];
    const int*   edges    = (const int*)d_in[1];
    const int*   batch    = (const int*)d_in[2];
    const float* W1 = (const float*)d_in[3];
    const float* b1 = (const float*)d_in[4];
    const float* k21 = (const float*)d_in[5];
    const float* W2 = (const float*)d_in[6];
    const float* b2 = (const float*)d_in[7];
    const float* k22 = (const float*)d_in[8];
    const float* g1 = (const float*)d_in[9];
    const float* be1 = (const float*)d_in[10];
    const float* m1 = (const float*)d_in[11];
    const float* v1 = (const float*)d_in[12];
    const float* g2 = (const float*)d_in[13];
    const float* be2 = (const float*)d_in[14];
    const float* m2 = (const float*)d_in[15];
    const float* v2 = (const float*)d_in[16];

    const int N  = in_sizes[0] / 128;
    const int E  = in_sizes[1] / 2;
    const int NB = in_sizes[2];
    const int* src = edges;
    const int* dst = edges + E;

    // ---- workspace layout (bytes, 256-aligned) ----
    char* ws = (char*)d_ws;
    auto al = [](size_t x) { return (x + 255) & ~(size_t)255; };
    size_t off = 0;
    float*  p_dinv = (float*)(ws + off);  off += al((size_t)N * 4);
    float*  p_coef = (float*)(ws + off);  off += al((size_t)E * 4);
    __bf16* p_af16 = (__bf16*)(ws + off); off += al((size_t)N * 128 * 2);  // bf16 features
    float*  p_h1   = (float*)(ws + off);  off += al((size_t)N * 64 * 4);
    float*  p_agg1 = (float*)(ws + off);  off += al((size_t)N * 64 * 4);
    __bf16* p_x1bf = (__bf16*)(ws + off); off += al((size_t)N * 64 * 2);   // bf16 x1
    __bf16* p_w1pk = (__bf16*)(ws + off); off += al((size_t)4 * 4 * 32 * 16 * 2);
    __bf16* p_w2pk = (__bf16*)(ws + off); off += al((size_t)2 * 3 * 32 * 16 * 2);
    float*  p_loss = (float*)(ws + off);  off += 256;
    // aliases (safe by stream ordering):
    float* p_h2   = p_h1;                 // h1 dead after epi1
    float* p_agg2 = p_agg1;               // agg1 dead after epi1
    float* p_x2   = (float*)p_af16;       // bf16 features dead after gemm1 (16MB <= 25.6MB)

    float* out = (float*)d_out;
    const int T = 256;

    // zero accumulators
    hipMemsetAsync(p_agg1, 0, (size_t)N * 64 * 4, stream);
    hipMemsetAsync(p_loss, 0, 4, stream);

    // degree / normalization (shared by both convs)
    k_deg_init<<<(N + T - 1) / T, T, 0, stream>>>(p_dinv, N);
    k_deg_acc<<<(E + T - 1) / T, T, 0, stream>>>(dst, p_dinv, E);
    k_rsqrt_inplace<<<(N + T - 1) / T, T, 0, stream>>>(p_dinv, N);
    k_coef<<<(E + T - 1) / T, T, 0, stream>>>(src, dst, p_dinv, p_coef, E);

    // precision prep: bf16 features + packed weights
    {
        long long npairs = (long long)N * 64;   // N*128/2
        k_cvt_bf16<<<(unsigned)((npairs + T - 1) / T), T, 0, stream>>>(
            features, (unsigned*)p_af16, npairs);
    }
    k_pack_w<4, 4, 64><<<2, T, 0, stream>>>(W1, p_w1pk);   // 512 threads
    k_pack_w<2, 3, 40><<<1, T, 0, stream>>>(W2, p_w2pk);   // 192 threads

    // conv1
    int nwaves = (N + 15) / 16;
    int blocks = (nwaves + 7) / 8;               // 8 waves / 256-thread block
    k_gemm1_wmma<<<blocks, T, 0, stream>>>(p_af16, p_w1pk, p_h1, N, nwaves);
    {
        long long total = (long long)E * 64;
        k_scatter<64><<<(unsigned)((total + T - 1) / T), T, 0, stream>>>(
            src, dst, p_coef, p_h1, p_agg1, total);
    }
    k_epi1<<<((N * 64 + T - 1) / T), T, 0, stream>>>(p_h1, p_agg1, p_dinv, b1, k21,
                                                     g1, be1, m1, v1, p_x1bf, p_loss, N);

    // conv2 (reuses aliased buffers)
    hipMemsetAsync(p_agg2, 0, (size_t)N * 40 * 4, stream);
    k_gemm2_wmma<<<blocks, T, 0, stream>>>(p_x1bf, p_w2pk, p_h2, N, nwaves);
    {
        long long total = (long long)E * 40;
        k_scatter<40><<<(unsigned)((total + T - 1) / T), T, 0, stream>>>(
            src, dst, p_coef, p_h2, p_agg2, total);
    }
    k_epi2<<<((N * 40 + T - 1) / T), T, 0, stream>>>(p_h2, p_agg2, p_dinv, b2, k22,
                                                     g2, be2, m2, v2, p_x2, N);

    // output: log_softmax rows + helm loss
    k_logsoftmax<<<(NB + T - 1) / T, T, 0, stream>>>(p_x2, batch, out, NB);
    k_finalize_loss<<<1, 32, 0, stream>>>(p_loss, out, NB * 40, 1.0f / ((float)N * 64.0f));
}
